// GRU_22720376995892
// MI455X (gfx1250) — matrix operand order repaired and checked by reading, not compiled, not run on hardware
//
#include <hip/hip_runtime.h>
#include <hip/hip_bf16.h>

// ---------------------------------------------------------------------------
// GRU step, B=512 H=1024 V=50000, fp32 in/out.
//   gi = embedded @ W_ih^T   (512 x 3072, K=50000)  dominant; split-K x4
//   gh = h @ W_hh^T          (512 x 3072, K=1024)
//   gates -> h_new           (512 x 1024)  (sums split-K partials of gi)
//   logit = tanh(h_new @ W_ho[target]^T + b_ho[target])   (512x512 gather-GEMM)
// GEMM: fp32 HBM -> regs (prefetch, double-buffered) -> cvt bf16 -> LDS
// (ping-pong) -> v_wmma_f32_16x16x32_bf16, f32 accumulators.
// ---------------------------------------------------------------------------

typedef __attribute__((ext_vector_type(16))) __bf16 v16bf;
typedef __attribute__((ext_vector_type(8)))  __bf16 v8bf;
typedef __attribute__((ext_vector_type(4)))  __bf16 v4bf;
typedef __attribute__((ext_vector_type(8)))  float  v8f;

#define BM 128
#define BN 128
#define BK 64
#define LDP 72   // padded LDS k-stride (elements): 144B rows, conflict-friendly

// C[m][n] = sum_{k in split} A[m][k] * Brows[n][k]
//   Brows[n] = Bm + (bidx ? bidx[n] : n)*ldb ; do_tanh: C = tanh(acc + bias[.])
__global__ __launch_bounds__(256) void gemm_bf16_wmma(
    const float* __restrict__ A, int lda,
    const float* __restrict__ Bm, int ldb,
    const long long* __restrict__ bidx,
    const float* __restrict__ bias,
    float* __restrict__ C, int ldc, size_t c_split_stride,
    int K, int nsplit, int do_tanh)
{
    __shared__ __bf16 As[2][BM * LDP];
    __shared__ __bf16 Bs[2][BN * LDP];

    const int tid   = threadIdx.x;
    const int lane  = tid & 31;
    const int wave  = tid >> 5;
    const int waveM = wave & 1;   // 2 waves over M: 64 rows each
    const int waveN = wave >> 1;  // 4 waves over N: 32 cols each
    const int lm    = lane & 15;
    const int lh    = lane >> 4;
    const int m0    = blockIdx.y * BM;
    const int n0    = blockIdx.x * BN;

    const int chunk = (K + nsplit - 1) / nsplit;   // multiple of 4 at call sites
    const int kbeg  = blockIdx.z * chunk;
    const int kend  = min(K, kbeg + chunk);
    C += (size_t)blockIdx.z * c_split_stride;

    // ---- loop-invariant staging geometry: 8 float4 per thread per matrix ----
    int rowIdx[8], kkOff[8];
    const float* arow[8];
    const float* brow[8];
    #pragma unroll
    for (int i = 0; i < 8; ++i) {
        int s     = i * 256 + tid;
        int row   = s >> 4;            // 0..127
        rowIdx[i] = row;
        kkOff[i]  = (s & 15) << 2;     // 0..60
        arow[i]   = A + (size_t)(m0 + row) * lda;
        brow[i]   = bidx ? (Bm + (size_t)bidx[n0 + row] * ldb)
                         : (Bm + (size_t)(n0 + row) * ldb);
    }

    float4 ra[8], rb[8];
    auto load_stage = [&](int kb) {
        #pragma unroll
        for (int i = 0; i < 8; ++i) {
            int k = kb + kkOff[i];
            float4 z4 = make_float4(0.f, 0.f, 0.f, 0.f);
            ra[i] = z4; rb[i] = z4;
            if (k < kend) {            // chunk % 4 == 0 at all call sites
                ra[i] = *(const float4*)(arow[i] + k);
                rb[i] = *(const float4*)(brow[i] + k);
            }
        }
    };
    auto store_stage = [&](int b) {
        #pragma unroll
        for (int i = 0; i < 8; ++i) {
            v4bf a4 = { (__bf16)ra[i].x, (__bf16)ra[i].y, (__bf16)ra[i].z, (__bf16)ra[i].w };
            v4bf b4 = { (__bf16)rb[i].x, (__bf16)rb[i].y, (__bf16)rb[i].z, (__bf16)rb[i].w };
            *(v4bf*)(&As[b][rowIdx[i] * LDP + kkOff[i]]) = a4;
            *(v4bf*)(&Bs[b][rowIdx[i] * LDP + kkOff[i]]) = b4;
        }
    };

    v8f acc[4][2] = {};   // 4 m-tiles x 2 n-tiles of 16x16 f32

    // ---- software pipeline: prefetch regs + ping-pong LDS, 1 barrier/stage --
    load_stage(kbeg);
    int buf = 0;
    for (int k0 = kbeg; k0 < kend; k0 += BK) {
        store_stage(buf);
        __syncthreads();
        if (k0 + BK < kend) load_stage(k0 + BK);   // HBM loads fly over WMMAs

        #pragma unroll
        for (int ks = 0; ks < 2; ++ks) {
            const int ko = ks * 32;
            v16bf afrag[4], bfrag[2];
            #pragma unroll
            for (int mt = 0; mt < 4; ++mt) {
                // A 16x32: lane half lh -> K chunks [lh*8,+8) and [16+lh*8,+8)
                const __bf16* p = &As[buf][(waveM * 64 + mt * 16 + lm) * LDP + ko + lh * 8];
                v8bf lo = *(const v8bf*)p;
                v8bf hi = *(const v8bf*)(p + 16);
                afrag[mt] = __builtin_shufflevector(lo, hi,
                    0,1,2,3,4,5,6,7,8,9,10,11,12,13,14,15);
            }
            #pragma unroll
            for (int nt = 0; nt < 2; ++nt) {
                // B 32x16: lane half lh -> K chunks [lh*8,+8) and [16+lh*8,+8)
                const __bf16* p = &Bs[buf][(waveN * 32 + nt * 16 + lm) * LDP + ko + lh * 8];
                v8bf lo = *(const v8bf*)p;
                v8bf hi = *(const v8bf*)(p + 16);
                bfrag[nt] = __builtin_shufflevector(lo, hi,
                    0,1,2,3,4,5,6,7,8,9,10,11,12,13,14,15);
            }
            #pragma unroll
            for (int mt = 0; mt < 4; ++mt)
                #pragma unroll
                for (int nt = 0; nt < 2; ++nt)
                    acc[mt][nt] = __builtin_amdgcn_wmma_f32_16x16x32_bf16(
                        false, afrag[mt], false, bfrag[nt],
                        (short)0, acc[mt][nt], false, false);
        }
        buf ^= 1;
    }

    // ---- epilogue: C/D layout — VGPR r: m = lh*8 + r, n = lm ----
    #pragma unroll
    for (int mt = 0; mt < 4; ++mt) {
        #pragma unroll
        for (int nt = 0; nt < 2; ++nt) {
            int col = n0 + waveN * 32 + nt * 16 + lm;
            float b = 0.f;
            if (do_tanh && bias) b = bias[bidx ? (int)bidx[col] : col];
            #pragma unroll
            for (int r = 0; r < 8; ++r) {
                int row = m0 + waveM * 64 + mt * 16 + lh * 8 + r;
                float v = acc[mt][nt][r];
                if (do_tanh) v = tanhf(v + b);
                C[(size_t)row * ldc + col] = v;
            }
        }
    }
}

// gates: sums gi split-K partials, then r,z = sigmoid; n = tanh;
// h_new = (1-z)*n + z*h
__global__ __launch_bounds__(256) void gru_gates(
    const float* __restrict__ gi, int nsplit,
    const float* __restrict__ gh,
    const float* __restrict__ b_ih, const float* __restrict__ b_hh,
    const float* __restrict__ h, float* __restrict__ h_new)
{
    int idx = blockIdx.x * 256 + threadIdx.x;    // 512*1024 total
    int i = idx >> 10, j = idx & 1023;
    float xr = 0.f, xz = 0.f, xn = 0.f;
    for (int s = 0; s < nsplit; ++s) {
        const float* g = gi + (size_t)s * (512 * 3072) + (size_t)i * 3072;
        xr += g[j];
        xz += g[1024 + j];
        xn += g[2048 + j];
    }
    const float* ghr = gh + (size_t)i * 3072;
    xr += b_ih[j]        + ghr[j]        + b_hh[j];
    xz += b_ih[1024 + j] + ghr[1024 + j] + b_hh[1024 + j];
    xn += b_ih[2048 + j];
    float hn = ghr[2048 + j] + b_hh[2048 + j];
    float r = 1.f / (1.f + __expf(-xr));
    float z = 1.f / (1.f + __expf(-xz));
    float n = tanhf(xn + r * hn);
    h_new[idx] = (1.f - z) * n + z * h[idx];
}

extern "C" void kernel_launch(void* const* d_in, const int* in_sizes, int n_in,
                              void* d_out, int out_size, void* d_ws, size_t ws_size,
                              hipStream_t stream) {
    const float*     embedded = (const float*)d_in[0];       // (512, 50000)
    const long long* target   = (const long long*)d_in[1];   // (512,) int64
    const float*     hidden   = (const float*)d_in[2];       // (1, 512, 1024)
    const float*     W_ih     = (const float*)d_in[3];       // (3072, 50000)
    const float*     W_hh     = (const float*)d_in[4];       // (3072, 1024)
    const float*     b_ih     = (const float*)d_in[5];       // (3072,)
    const float*     b_hh     = (const float*)d_in[6];       // (3072,)
    const float*     W_ho     = (const float*)d_in[7];       // (50000, 1024)
    const float*     b_ho     = (const float*)d_in[8];       // (50000,)

    float* logit = (float*)d_out;               // (512, 512)
    float* h_new = logit + 512 * 512;           // (512, 1024)

    const size_t GI = (size_t)512 * 3072;       // one gi partial (elements)
    // split-K x4 for the dominant GEMM if workspace allows (deterministic)
    int nsplit = (ws_size >= (4 * GI + GI) * sizeof(float)) ? 4 : 1;
    float* gi = (float*)d_ws;                   // nsplit x (512, 3072)
    float* gh = gi + (size_t)nsplit * GI;       // (512, 3072)

    dim3 blk(256);

    // gi[z] = embedded @ W_ih^T (K-range z)    M=512 N=3072 K=50000
    gemm_bf16_wmma<<<dim3(3072 / BN, 512 / BM, nsplit), blk, 0, stream>>>(
        embedded, 50000, W_ih, 50000, nullptr, nullptr,
        gi, 3072, GI, 50000, nsplit, 0);

    // gh = h @ W_hh^T                          M=512 N=3072 K=1024
    gemm_bf16_wmma<<<dim3(3072 / BN, 512 / BM, 1), blk, 0, stream>>>(
        hidden, 1024, W_hh, 1024, nullptr, nullptr,
        gh, 3072, 0, 1024, 1, 0);

    // gates -> h_new (written directly into its d_out slot)
    gru_gates<<<(512 * 1024) / 256, blk, 0, stream>>>(
        gi, nsplit, gh, b_ih, b_hh, hidden, h_new);

    // logit = tanh(h_new @ W_ho[target]^T + b_ho[target])  M=512 N=512 K=1024
    gemm_bf16_wmma<<<dim3(512 / BN, 512 / BM, 1), blk, 0, stream>>>(
        h_new, 1024, W_ho, 1024, target, b_ho,
        logit, 512, 0, 1024, 1, 1);
}
